// SessionGraph_49847390437917
// MI455X (gfx1250) — matrix operand order repaired and checked by reading, not compiled, not run on hardware
//
#include <hip/hip_runtime.h>
#include <hip/hip_bf16.h>
#include <math.h>

// Problem constants (match reference)
#define Bn 128
#define Sn 256
#define Hn 128
#define Mn (Bn * Sn)   // 32768 rows for flattened [B,S,*] GEMMs

typedef __attribute__((ext_vector_type(16))) __bf16 v16bf;
typedef __attribute__((ext_vector_type(8)))  float  v8f;
typedef __attribute__((ext_vector_type(4)))  float  f32x4;

#define LP 24   // padded per-lane fragment stride in bf16 elems (48 B, 16B-aligned)

// -----------------------------------------------------------------------------
// Tiled GEMM, bf16 WMMA, f32 accumulate.  C[M,N] = A[M,K]*B(k,n) + bias[n]
//   A: f32 row-major (lda, batch strideA)
//   btrans==1: Bsrc is [N,K] row-major (weights; B(k,n)=Bsrc[n*ldb+k])
//   btrans==0: Bsrc is [K,N] row-major (B(k,n)=Bsrc[k*ldb+n])
// Block: 256 threads = 8 wave32. C tile 128x64; each wave computes a 32x32
// patch = 4 x v_wmma_f32_16x16x32_bf16 per K step (2 A-frags x 2 B-frags).
// LDS tiles are stored PRE-SWIZZLED in WMMA fragment order so fragment loads
// are 2x ds_load_b128 per fragment and staging stores are 1x ds_store_b128.
// Requires M%128==0, N%64==0, K%32==0 (true for all GEMMs here).
//
// Fragment layouts used here (wave32, 16-bit; same contiguous k order for A and B):
//   A 16x32: lane L holds row M=L&15; elem e -> K = (L>>4)*16 + e
//            => halves e=0..7 / e=8..15 are K-contiguous runs of 8.
//   B 32x16: lane L holds col N=L&15; elem e -> K = (L>>4)*16 + e
//            => halves are K-contiguous runs of 8.
// -----------------------------------------------------------------------------
__global__ __launch_bounds__(256)
void gemm_bf16_wmma(const float* __restrict__ A, long long lda, long long strideA,
                    const float* __restrict__ Bsrc, long long ldb, long long strideB,
                    int btrans,
                    const float* __restrict__ bias,
                    float* __restrict__ C, long long ldc, long long strideC,
                    int M, int N, int K)
{
    __shared__ __align__(16) __bf16 sA[8][32][LP];  // 8 row-tiles of 16
    __shared__ __align__(16) __bf16 sB[4][32][LP];  // 4 col-tiles of 16

    const int tid  = threadIdx.x;
    const int lane = tid & 31;
    const int wave = tid >> 5;
    const int rl   = lane & 15;
    const int rsel = lane >> 4;

    const int row0 = blockIdx.y * 128;
    const int col0 = blockIdx.x * 64;

    const float* Ab = A    + (long long)blockIdx.z * strideA;
    const float* Bb = Bsrc + (long long)blockIdx.z * strideB;
    float*       Cb = C    + (long long)blockIdx.z * strideC;

    // ---- A staging: 512 slots (wr,lane,half), thread owns slot tid and tid+256.
    // slot s: wr=s>>6, ln=(s>>1)&31, half=s&1.
    // source row = row0 + wr*16 + (ln&15); k-run = (ln>>4)*16 + (half<<3) + [0..7]
    const int wrA0 = tid >> 6,           lnA0 = (tid >> 1) & 31,  hA0 = tid & 1;
    const int sA1i = tid + 256;
    const int wrA1 = sA1i >> 6,          lnA1 = (sA1i >> 1) & 31, hA1 = sA1i & 1;
    const float* pA0 = Ab + (long long)(row0 + wrA0 * 16 + (lnA0 & 15)) * lda
                          + (hA0 << 3) + ((lnA0 >> 4) << 4);
    const float* pA1 = Ab + (long long)(row0 + wrA1 * 16 + (lnA1 & 15)) * lda
                          + (hA1 << 3) + ((lnA1 >> 4) << 4);
    __bf16* stA0 = &sA[wrA0][lnA0][hA0 * 8];
    __bf16* stA1 = &sA[wrA1][lnA1][hA1 * 8];

    // ---- B staging
    const float* pB;
    long long    pBstep;
    __bf16*      stB = nullptr;           // btrans path: wide store target
    const int    rB  = tid >> 3;          // non-trans path: k-row
    const int    cgB = (tid & 7) << 3;    // non-trans path: col group
    if (btrans) {
        // 256 slots (ct,ln,half): col = col0+ct*16+(ln&15); k-run = (ln>>4)*16+half*8+[0..7]
        int ct = tid >> 6, ln = (tid >> 1) & 31, hf = tid & 1;
        pB     = Bb + (long long)(col0 + ct * 16 + (ln & 15)) * ldb
                    + ((ln >> 4) << 4) + (hf << 3);
        stB    = &sB[ct][ln][hf * 8];
        pBstep = 32;
    } else {
        // thread owns row rB, cols col0+cgB..+7 (contiguous global read)
        pB     = Bb + (long long)rB * ldb + col0 + cgB;
        pBstep = 32 * ldb;
    }

    // ---- per-wave fragment sources (loop invariant)
    const int g   = wave & 3;     // row group: row tiles 2g, 2g+1
    const int cgw = wave >> 2;    // col group: col tiles 2cgw, 2cgw+1
    const __bf16* fAp0 = &sA[2 * g][lane][0];
    const __bf16* fAp1 = &sA[2 * g + 1][lane][0];
    const __bf16* fBp0 = &sB[2 * cgw][lane][0];
    const __bf16* fBp1 = &sB[2 * cgw + 1][lane][0];

    v8f acc00 = {}, acc01 = {}, acc10 = {}, acc11 = {};

    for (int k0 = 0; k0 < K; k0 += 32) {
        // wide global loads (all 16B-aligned)
        f32x4 a00 = *(const f32x4*)(pA0);
        f32x4 a01 = *(const f32x4*)(pA0 + 4);
        f32x4 a10 = *(const f32x4*)(pA1);
        f32x4 a11 = *(const f32x4*)(pA1 + 4);
        f32x4 b0  = *(const f32x4*)(pB);
        f32x4 b1  = *(const f32x4*)(pB + 4);
        pA0 += 32; pA1 += 32; pB += pBstep;

        // convert + wide LDS stores (A)
        __bf16 ta0[8], ta1[8];
        #pragma unroll
        for (int i = 0; i < 4; i++) {
            ta0[i]     = (__bf16)a00[i];
            ta0[i + 4] = (__bf16)a01[i];
            ta1[i]     = (__bf16)a10[i];
            ta1[i + 4] = (__bf16)a11[i];
        }
        __builtin_memcpy(stA0, ta0, 16);
        __builtin_memcpy(stA1, ta1, 16);

        // convert + store (B)
        __bf16 tb[8];
        #pragma unroll
        for (int i = 0; i < 4; i++) {
            tb[i]     = (__bf16)b0[i];
            tb[i + 4] = (__bf16)b1[i];
        }
        if (btrans) {
            __builtin_memcpy(stB, tb, 16);
        } else {
            // scatter into fragment order: (k=rB, col c) -> sB[c>>4][(rB>>4)*16 + (c&15)][rB&15]
            #pragma unroll
            for (int i = 0; i < 8; i++) {
                int c = cgB + i;
                sB[c >> 4][((rB >> 4) << 4) + (c & 15)][rB & 15] = tb[i];
            }
        }
        __syncthreads();

        // fragment loads: 2x ds_load_b128 each
        v16bf fa0, fa1, fb0, fb1;
        __builtin_memcpy(&fa0, fAp0, 32);
        __builtin_memcpy(&fa1, fAp1, 32);
        __builtin_memcpy(&fb0, fBp0, 32);
        __builtin_memcpy(&fb1, fBp1, 32);

        acc00 = __builtin_amdgcn_wmma_f32_16x16x32_bf16(false, fa0, false, fb0, (short)0, acc00, false, false);
        acc01 = __builtin_amdgcn_wmma_f32_16x16x32_bf16(false, fa0, false, fb1, (short)0, acc01, false, false);
        acc10 = __builtin_amdgcn_wmma_f32_16x16x32_bf16(false, fa1, false, fb0, (short)0, acc10, false, false);
        acc11 = __builtin_amdgcn_wmma_f32_16x16x32_bf16(false, fa1, false, fb1, (short)0, acc11, false, false);
        __syncthreads();
    }

    // epilogue: C/D layout -> VGPR v: M = v + 8*rsel, N = lane&15
    v8f accs[2][2] = {{acc00, acc01}, {acc10, acc11}};
    #pragma unroll
    for (int ri = 0; ri < 2; ri++) {
        #pragma unroll
        for (int ci = 0; ci < 2; ci++) {
            int n = col0 + (2 * cgw + ci) * 16 + rl;
            float bv = bias[n];
            #pragma unroll
            for (int v = 0; v < 8; v++) {
                int r = row0 + (2 * g + ri) * 16 + v + 8 * rsel;
                Cb[(long long)r * ldc + n] = accs[ri][ci][v] + bv;
            }
        }
    }
}

// -----------------------------------------------------------------------------
// GRU gate elementwise: newh = h - sigmoid(i_i+h_i)*(h - tanh(i_n + r*h_n))
// -----------------------------------------------------------------------------
__global__ __launch_bounds__(256)
void gru_kernel(const float* __restrict__ gi, const float* __restrict__ gh,
                const float* __restrict__ hidden, float* __restrict__ newh)
{
    long long idx = (long long)blockIdx.x * 256 + threadIdx.x;
    if (idx >= (long long)Mn * Hn) return;
    int m = (int)(idx >> 7);
    int c = (int)(idx & 127);
    const float* gim = gi + (long long)m * 384;
    const float* ghm = gh + (long long)m * 384;
    float h  = hidden[idx];
    float r  = 1.f / (1.f + expf(-(gim[c]       + ghm[c])));
    float ig = 1.f / (1.f + expf(-(gim[128 + c] + ghm[128 + c])));
    float ng = tanhf(gim[256 + c] + r * ghm[256 + c]);
    newh[idx] = h - ig * (h - ng);
}

// -----------------------------------------------------------------------------
// q0 = hidden[b, last_idx_b, :] @ W_q0.T + b_q0
// -----------------------------------------------------------------------------
__global__ __launch_bounds__(128)
void q0_kernel(const float* __restrict__ newh, const int* __restrict__ mask,
               const float* __restrict__ Wq0, const float* __restrict__ bq0,
               float* __restrict__ q0out)
{
    int b = blockIdx.x, n = threadIdx.x;
    __shared__ int   mred[128];
    __shared__ float ht[128];
    mred[n] = mask[b * Sn + n] + mask[b * Sn + 128 + n];
    __syncthreads();
    for (int s = 64; s > 0; s >>= 1) {
        if (n < s) mred[n] += mred[n + s];
        __syncthreads();
    }
    int last = mred[0] - 1;
    ht[n] = newh[((long long)b * Sn + last) * Hn + n];
    __syncthreads();
    float a = bq0[n];
    #pragma unroll 4
    for (int k = 0; k < 128; k++) a += ht[k] * Wq0[n * 128 + k];
    q0out[b * Hn + n] = a;
}

// -----------------------------------------------------------------------------
// Collapsed attention readout (score is independent of the query index):
//   out[b,:] = (sum_i mask_i*[mask_i>0]) *
//              sum_j softmax_h(2*softmax_j(sigmoid(q0.q1_j)))[h,j] * q2[b,j,h,:]
// -----------------------------------------------------------------------------
__global__ __launch_bounds__(256)
void attn_kernel(const float* __restrict__ q0, const float* __restrict__ q1,
                 const float* __restrict__ q2, const int* __restrict__ mask,
                 float* __restrict__ out)
{
    const int b = blockIdx.x, j = threadIdx.x;
    __shared__ float red[256];
    __shared__ float accs[128];

    const float* q0p = q0 + (long long)b * Hn;
    const float* q1p = q1 + ((long long)b * Sn + j) * Hn;
    const float* q2p = q2 + ((long long)b * Sn + j) * Hn;

    float sc[8];
    #pragma unroll
    for (int h = 0; h < 8; h++) {
        float s = 0.f;
        #pragma unroll
        for (int d = 0; d < 16; d++) s += q0p[h * 16 + d] * q1p[h * 16 + d];
        sc[h] = 1.f / (1.f + expf(-s));
    }

    // inner softmax over j (sigmoid outputs bounded: no max-shift needed)
    float psum[8];
    #pragma unroll
    for (int h = 0; h < 8; h++) {
        red[j] = expf(sc[h]);
        __syncthreads();
        for (int s = 128; s > 0; s >>= 1) {
            if (j < s) red[j] += red[j + s];
            __syncthreads();
        }
        psum[h] = red[0];
        __syncthreads();
    }

    float w[8], denom = 0.f;
    #pragma unroll
    for (int h = 0; h < 8; h++) {
        float p = expf(sc[h]) / psum[h];
        w[h] = expf(2.f * p);
        denom += w[h];
    }
    float inv = 1.f / denom;

    if (j < 128) accs[j] = 0.f;
    __syncthreads();
    #pragma unroll
    for (int h = 0; h < 8; h++) {
        float wh = w[h] * inv;
        #pragma unroll
        for (int d = 0; d < 16; d++)
            atomicAdd(&accs[h * 16 + d], wh * q2p[h * 16 + d]);
    }

    int mv = mask[b * Sn + j];
    red[j] = (mv > 0) ? (float)mv : 0.f;
    __syncthreads();
    for (int s = 128; s > 0; s >>= 1) {
        if (j < s) red[j] += red[j + s];
        __syncthreads();
    }
    if (j < 128) out[(long long)b * Hn + j] = accs[j] * red[0];
}

// -----------------------------------------------------------------------------
extern "C" void kernel_launch(void* const* d_in, const int* in_sizes, int n_in,
                              void* d_out, int out_size, void* d_ws, size_t ws_size,
                              hipStream_t stream)
{
    const float* A      = (const float*)d_in[0];   // [B,S,2S]
    const float* hidden = (const float*)d_in[1];   // [B,S,H]
    const int*   mask   = (const int*)  d_in[2];   // [B,S]
    const float* w_ih   = (const float*)d_in[3];   // [3H,2H]
    const float* w_hh   = (const float*)d_in[4];   // [3H,H]
    const float* b_ih   = (const float*)d_in[5];
    const float* b_hh   = (const float*)d_in[6];
    const float* b_iah  = (const float*)d_in[7];
    const float* b_oah  = (const float*)d_in[8];
    const float* W_ein  = (const float*)d_in[9];
    const float* b_ein  = (const float*)d_in[10];
    const float* W_eout = (const float*)d_in[11];
    const float* b_eout = (const float*)d_in[12];
    const float* W_q0   = (const float*)d_in[13];
    const float* b_q0   = (const float*)d_in[14];
    const float* W_q1   = (const float*)d_in[15];
    const float* b_q1   = (const float*)d_in[16];
    const float* W_q2   = (const float*)d_in[17];
    const float* b_q2   = (const float*)d_in[18];

    // Workspace: he_in | he_out | msgs(2H) | gi(3H) | gh(3H) | newh  (~184 MiB f32)
    float* ws = (float*)d_ws;
    const size_t MH = (size_t)Mn * Hn;
    float* he_in  = ws;
    float* he_out = he_in  + MH;
    float* msgs   = he_out + MH;
    float* gi     = msgs   + 2 * MH;
    float* gh     = gi     + 3 * MH;
    float* newh   = gh     + 3 * MH;
    float* q1buf  = he_in;             // reuse: he_* dead after adjacency GEMMs
    float* q2buf  = he_out;
    float* q0buf  = msgs;              // reuse: msgs dead after gi GEMM

    dim3 blk(256);

    // 1-2: he_in/out = hidden @ W_e{in,out}.T + b
    gemm_bf16_wmma<<<dim3(Hn / 64, Mn / 128, 1), blk, 0, stream>>>(
        hidden, Hn, 0, W_ein, Hn, 0, 1, b_ein, he_in, Hn, 0, Mn, Hn, Hn);
    gemm_bf16_wmma<<<dim3(Hn / 64, Mn / 128, 1), blk, 0, stream>>>(
        hidden, Hn, 0, W_eout, Hn, 0, 1, b_eout, he_out, Hn, 0, Mn, Hn, Hn);

    // 3-4: msgs = [A_in @ he_in + b_iah | A_out @ he_out + b_oah] (batched over B)
    gemm_bf16_wmma<<<dim3(Hn / 64, Sn / 128, Bn), blk, 0, stream>>>(
        A, 2 * Sn, (long long)Sn * 2 * Sn,
        he_in, Hn, (long long)Sn * Hn, 0, b_iah,
        msgs, 2 * Hn, (long long)Sn * 2 * Hn, Sn, Hn, Sn);
    gemm_bf16_wmma<<<dim3(Hn / 64, Sn / 128, Bn), blk, 0, stream>>>(
        A + Sn, 2 * Sn, (long long)Sn * 2 * Sn,
        he_out, Hn, (long long)Sn * Hn, 0, b_oah,
        msgs + Hn, 2 * Hn, (long long)Sn * 2 * Hn, Sn, Hn, Sn);

    // 5: gi = msgs @ w_ih.T + b_ih
    gemm_bf16_wmma<<<dim3(384 / 64, Mn / 128, 1), blk, 0, stream>>>(
        msgs, 2 * Hn, 0, w_ih, 2 * Hn, 0, 1, b_ih, gi, 3 * Hn, 0, Mn, 3 * Hn, 2 * Hn);
    // 6: gh = hidden @ w_hh.T + b_hh
    gemm_bf16_wmma<<<dim3(384 / 64, Mn / 128, 1), blk, 0, stream>>>(
        hidden, Hn, 0, w_hh, Hn, 0, 1, b_hh, gh, 3 * Hn, 0, Mn, 3 * Hn, Hn);

    // 7: GRU elementwise
    gru_kernel<<<(Mn * Hn + 255) / 256, blk, 0, stream>>>(gi, gh, hidden, newh);

    // 8-9: q1/q2 = newh @ W_q{1,2}.T + b
    gemm_bf16_wmma<<<dim3(Hn / 64, Mn / 128, 1), blk, 0, stream>>>(
        newh, Hn, 0, W_q1, Hn, 0, 1, b_q1, q1buf, Hn, 0, Mn, Hn, Hn);
    gemm_bf16_wmma<<<dim3(Hn / 64, Mn / 128, 1), blk, 0, stream>>>(
        newh, Hn, 0, W_q2, Hn, 0, 1, b_q2, q2buf, Hn, 0, Mn, Hn, Hn);

    // 10: q0 from last valid row per batch
    q0_kernel<<<Bn, 128, 0, stream>>>(newh, mask, W_q0, b_q0, q0buf);

    // 11: attention readout -> d_out [B,H]
    attn_kernel<<<Bn, 256, 0, stream>>>(q0buf, q1buf, q2buf, mask, (float*)d_out);
}
